// SparseCoding_16741782520299
// MI455X (gfx1250) — matrix operand order repaired and checked by reading, not compiled, not run on hardware
//
#include <hip/hip_runtime.h>

// ---------------------------------------------------------------------------
// ISTA sparse coding, restructured as x <- soft(x*A + b) with
//   A = I - 2*LR*W*W^T   (256x256, SYMMETRIC, bf16, LDS-resident)
//   b = 2*LR*in*W^T      (per-row-tile, register-resident f32)
// Step GEMMs run on v_wmma_f32_16x16x32_bf16 (wave32).
// All WMMA operand fragments are fed from bank-conflict-free padded LDS.
// ---------------------------------------------------------------------------

#define BATCH   131072
#define IDIM    128
#define ODIM    256
#define LRV     0.001f
#define THR     0.0001f          /* LR * GAMMA */
#define TWOLR   0.002f           /* 2 * LR     */
#define ASTRIDE 264              /* 256 + 8 bf16 pad -> 528B rows, +4 banks/row */
#define WSTRIDE 136              /* 128 + 8 bf16 pad -> 272B rows, +4 banks/row */
#define ROWS_PER_BLOCK 128
#define A_LDS_HW (ODIM * ASTRIDE)            /* 67584 hw = 135168 B */
#define X_LDS_HW (ROWS_PER_BLOCK * ASTRIDE)  /* 33792 hw =  67584 B */
#define W_LDS_HW (ODIM * WSTRIDE)            /* 34816 hw =  69632 B */
#define SMEM_BYTES ((A_LDS_HW + X_LDS_HW + W_LDS_HW) * 2)  /* 272384 B < 320 KB */

typedef __attribute__((ext_vector_type(16))) __bf16        v16bf;
typedef __attribute__((ext_vector_type(8)))  float         v8f;
typedef __attribute__((ext_vector_type(4)))  unsigned int  uint4v;
typedef __attribute__((ext_vector_type(2)))  unsigned int  uint2v;
typedef __attribute__((ext_vector_type(4)))  float         float4v;

struct __attribute__((aligned(16))) Pack32 { uint4v a, b; };

// Native f32 -> bf16 convert (backend picks v_cvt_*bf16_f32 on gfx1250).
__device__ __forceinline__ unsigned short f2bf(float f) {
  __bf16 h = (__bf16)f;
  return __builtin_bit_cast(unsigned short, h);
}

__device__ __forceinline__ v16bf load16x2(const void* p0, const void* p1) {
  Pack32 t;
  t.a = *(const uint4v*)p0;
  t.b = *(const uint4v*)p1;
  return __builtin_bit_cast(v16bf, t);
}

// A-operand fragment (16x32 bf16, M x K) from a row-major bf16 LDS tile.
// lanes 0-15: M=lane, K = kb+0..7 (V0-3) and kb+16..23 (V4-7)
// lanes 16-31: M=lane-16, K = kb+8..15 and kb+24..31
__device__ __forceinline__ v16bf load_afrag(const unsigned short* rows0,
                                            int lane, int k_base) {
  int m  = lane & 15;
  int k0 = k_base + ((lane & 16) ? 8 : 0);
  const unsigned short* p = rows0 + m * ASTRIDE + k0;
  return load16x2(p, p + 16);           // +16 halfwords = +32B (K+16)
}

// B-operand fragment (32x16 bf16, K x N) from an LDS matrix M that stores the
// TRANSPOSE of the B operand row-major (element (k,n) = M[n][k], contiguous
// along k).  For the step GEMM M = A (symmetric, so A itself works).
// lanes 0-15: N=lane, K=kb+0..7, kb+16..23 ; lanes 16-31: N=lane-16, K=kb+8..15, kb+24..31
__device__ __forceinline__ v16bf load_bfrag_lds(const unsigned short* M, int stride,
                                                int lane, int n_base, int k_base) {
  int n = n_base + (lane & 15);
  int k = k_base + ((lane & 16) ? 8 : 0);
  const unsigned short* p = M + n * stride + k;
  return load16x2(p, p + 16);           // +16 halfwords = +32B (K+16)
}

__device__ __forceinline__ v8f wmma_bf16(v16bf a, v16bf b, v8f c) {
  return __builtin_amdgcn_wmma_f32_16x16x32_bf16(
      /*neg_a=*/false, a, /*neg_b=*/false, b,
      /*c_mod=*/(short)0, c, /*reuse_a=*/false, /*reuse_b=*/false);
}

__device__ __forceinline__ v8f soft_thresh(v8f x) {
  v8f y;
#pragma unroll
  for (int i = 0; i < 8; ++i)
    y[i] = fmaxf(x[i] - THR, 0.0f) + fminf(x[i] + THR, 0.0f);
  return y;
}

// Store a 16x16 f32 C tile as bf16 into the wave's row-major LDS tile.
// C layout: lanes 0-15: N=lane, VGPR v -> M=v ; lanes 16-31: M=v+8.
__device__ __forceinline__ void store_xtile(unsigned short* rows0, int lane,
                                            int n_base, v8f y) {
  int n  = n_base + (lane & 15);
  int mb = (lane & 16) ? 8 : 0;
#pragma unroll
  for (int v = 0; v < 8; ++v)
    rows0[(mb + v) * ASTRIDE + n] = f2bf(y[v]);
}

// ---------------------------------------------------------------------------
// Prep: A = I - 2LR*W*W^T (bf16), W (bf16, [256][128]), W^T (bf16, [128][256])
// grid = 256 blocks (one per output row o1), 256 threads (one per o2).
// ---------------------------------------------------------------------------
__global__ void ista_prep_kernel(const float* __restrict__ W,
                                 unsigned short* __restrict__ Abf,
                                 unsigned short* __restrict__ Wbf,
                                 unsigned short* __restrict__ WTbf) {
  int o1 = blockIdx.x;
  int o2 = threadIdx.x;
  if (o2 < IDIM) {
    unsigned short h = f2bf(W[o1 * IDIM + o2]);
    Wbf[o1 * IDIM + o2]  = h;
    WTbf[o2 * ODIM + o1] = h;
  }
  const float* r1 = W + o1 * IDIM;
  const float* r2 = W + o2 * IDIM;
  float s = 0.0f;
#pragma unroll 8
  for (int i = 0; i < IDIM; ++i) s = fmaf(r1[i], r2[i], s);
  float a = (o1 == o2 ? 1.0f : 0.0f) - 2.0f * LRV * s;
  Abf[o1 * ODIM + o2] = f2bf(a);
}

// ---------------------------------------------------------------------------
// Main: per block, 128 rows of the batch; 8 waves x 16 rows each.
// ---------------------------------------------------------------------------
__global__ void __launch_bounds__(256)
ista_main_kernel(const float* __restrict__ inputs,
                 const unsigned short* __restrict__ Abf,
                 const unsigned short* __restrict__ Wbf,
                 const unsigned short* __restrict__ WTbf,
                 float* __restrict__ out) {
  extern __shared__ char smem[];
  unsigned short* A_lds = (unsigned short*)smem;                 // [256][ASTRIDE]
  unsigned short* x_lds = A_lds + A_LDS_HW;                      // [128][ASTRIDE]
  unsigned short* W_lds = x_lds + X_LDS_HW;                      // [256][WSTRIDE]

  const int tid  = threadIdx.x;
  const int lane = tid & 31;
  const int w    = tid >> 5;
  const int rowbase = blockIdx.x * ROWS_PER_BLOCK;

  // --- stage A (symmetric, bf16) into padded LDS: thread tid copies row tid
  {
    const uint4v* src = (const uint4v*)(Abf + tid * ODIM);
    uint4v* dst = (uint4v*)(A_lds + tid * ASTRIDE);
#pragma unroll 8
    for (int i = 0; i < 32; ++i) dst[i] = src[i];
  }
  // --- stage W (bf16 [256][128]) into padded LDS: thread tid copies row tid
  {
    const uint4v* src = (const uint4v*)(Wbf + tid * IDIM);
    uint4v* dst = (uint4v*)(W_lds + tid * WSTRIDE);
#pragma unroll 8
    for (int i = 0; i < 16; ++i) dst[i] = src[i];
  }
  // --- stage inputs (f32 -> bf16) into x_lds cols [0,128)
  for (int idx = tid; idx < ROWS_PER_BLOCK * (IDIM / 4); idx += 256) {
    int r  = idx >> 5;              // row 0..127
    int c4 = idx & 31;              // float4 index 0..31
    float4v f = ((const float4v*)(inputs + (rowbase + r) * IDIM))[c4];
    uint2v p;
    p.x = (unsigned int)f2bf(f.x) | ((unsigned int)f2bf(f.y) << 16);
    p.y = (unsigned int)f2bf(f.z) | ((unsigned int)f2bf(f.w) << 16);
    *(uint2v*)(x_lds + r * ASTRIDE + c4 * 4) = p;
  }
  __syncthreads();

  unsigned short* xw = x_lds + w * 16 * ASTRIDE;   // this wave's 16 rows

  // --- b = 2LR * in * W^T  (K=128); kept in registers, then x1 = soft(b)
  v8f breg[16];
#pragma unroll
  for (int nt = 0; nt < 16; nt += 2) {
    v8f acc0 = {0.f,0.f,0.f,0.f,0.f,0.f,0.f,0.f};
    v8f acc1 = {0.f,0.f,0.f,0.f,0.f,0.f,0.f,0.f};
#pragma unroll
    for (int kf = 0; kf < 4; ++kf) {
      v16bf a  = load_afrag(xw, lane, kf * 32);
      v16bf b0 = load_bfrag_lds(W_lds, WSTRIDE, lane, nt * 16,       kf * 32);
      v16bf b1 = load_bfrag_lds(W_lds, WSTRIDE, lane, (nt + 1) * 16, kf * 32);
      acc0 = wmma_bf16(a, b0, acc0);
      acc1 = wmma_bf16(a, b1, acc1);
    }
#pragma unroll
    for (int i = 0; i < 8; ++i) { acc0[i] *= TWOLR; acc1[i] *= TWOLR; }
    breg[nt]     = acc0;
    breg[nt + 1] = acc1;
  }
  // x1 = soft(b): each wave writes only its own rows -> no barrier needed.
#pragma unroll
  for (int nt = 0; nt < 16; ++nt)
    store_xtile(xw, lane, nt * 16, soft_thresh(breg[nt]));

  // --- 9 remaining ISTA steps: x <- soft(x*A + b), A symmetric in LDS.
  // Hot loop is wave-private (own rows only): zero barriers.
  // breg[nt] enters each accumulation chain as the C operand of the FIRST
  // WMMA (D != C), so no v_mov copies of breg are needed per iteration.
  for (int it = 0; it < 9; ++it) {
    v16bf xf[8];
#pragma unroll
    for (int kf = 0; kf < 8; ++kf) xf[kf] = load_afrag(xw, lane, kf * 32);
#pragma unroll
    for (int nt = 0; nt < 16; nt += 2) {
      v8f acc0 = wmma_bf16(xf[0],
                           load_bfrag_lds(A_lds, ASTRIDE, lane, nt * 16, 0),
                           breg[nt]);
      v8f acc1 = wmma_bf16(xf[0],
                           load_bfrag_lds(A_lds, ASTRIDE, lane, (nt + 1) * 16, 0),
                           breg[nt + 1]);
#pragma unroll
      for (int kf = 1; kf < 8; ++kf) {
        v16bf b0 = load_bfrag_lds(A_lds, ASTRIDE, lane, nt * 16,       kf * 32);
        v16bf b1 = load_bfrag_lds(A_lds, ASTRIDE, lane, (nt + 1) * 16, kf * 32);
        acc0 = wmma_bf16(xf[kf], b0, acc0);
        acc1 = wmma_bf16(xf[kf], b1, acc1);
      }
      store_xtile(xw, lane, nt * 16,       soft_thresh(acc0));
      store_xtile(xw, lane, (nt + 1) * 16, soft_thresh(acc1));
    }
  }

  // --- A is dead now: repurpose its LDS for W^T ([128][256] bf16, 264-stride)
  __syncthreads();                         // all waves done reading A
  {
    int r = tid >> 1;                      // row 0..127
    int part = (tid & 1) * 8;              // quarter-row: 8 uint4 = 64 halfwords
    const uint4v* src = (const uint4v*)(WTbf + r * ODIM) + part;
    uint4v* dst = (uint4v*)(A_lds + r * ASTRIDE) + part;
#pragma unroll
    for (int i = 0; i < 8; ++i) dst[i] = src[i];
  }
  __syncthreads();                         // W^T staged for everyone

  // --- reconstruction: out = x @ W  (K=256, N=128), B operand from W^T in LDS
  {
    v16bf xf[8];
#pragma unroll
    for (int kf = 0; kf < 8; ++kf) xf[kf] = load_afrag(xw, lane, kf * 32);
#pragma unroll
    for (int nt = 0; nt < 8; nt += 2) {
      v8f acc0 = {0.f,0.f,0.f,0.f,0.f,0.f,0.f,0.f};
      v8f acc1 = {0.f,0.f,0.f,0.f,0.f,0.f,0.f,0.f};
#pragma unroll
      for (int kf = 0; kf < 8; ++kf) {
        v16bf b0 = load_bfrag_lds(A_lds, ASTRIDE, lane, nt * 16,       kf * 32);
        v16bf b1 = load_bfrag_lds(A_lds, ASTRIDE, lane, (nt + 1) * 16, kf * 32);
        acc0 = wmma_bf16(xf[kf], b0, acc0);
        acc1 = wmma_bf16(xf[kf], b1, acc1);
      }
      int n0   = nt * 16 + (lane & 15);
      int mrow = rowbase + w * 16 + ((lane & 16) ? 8 : 0);
#pragma unroll
      for (int v = 0; v < 8; ++v) {
        out[(mrow + v) * IDIM + n0]      = acc0[v];
        out[(mrow + v) * IDIM + n0 + 16] = acc1[v];
      }
    }
  }
}

extern "C" void kernel_launch(void* const* d_in, const int* in_sizes, int n_in,
                              void* d_out, int out_size, void* d_ws, size_t ws_size,
                              hipStream_t stream) {
  const float* inputs = (const float*)d_in[0];   // [131072, 128] f32
  const float* W      = (const float*)d_in[1];   // [256, 128]    f32
  float* out          = (float*)d_out;           // [131072, 128] f32

  // workspace: A bf16 (128KB) + W bf16 (64KB) + W^T bf16 (64KB) = 256KB
  unsigned short* Abf  = (unsigned short*)d_ws;
  unsigned short* Wbf  = Abf + ODIM * ODIM;
  unsigned short* WTbf = Wbf + ODIM * IDIM;

  ista_prep_kernel<<<ODIM, 256, 0, stream>>>(W, Abf, Wbf, WTbf);
  ista_main_kernel<<<BATCH / ROWS_PER_BLOCK, 256, SMEM_BYTES, stream>>>(
      inputs, Abf, Wbf, WTbf, out);
}
